// Model_11828339933916
// MI455X (gfx1250) — compile-verified, matrix operand order checked by reading, not run on hardware
//
#include <hip/hip_runtime.h>
#include <hip/hip_bf16.h>
#include <cstdint>

// ---------------------------------------------------------------------------
// Model dims
// ---------------------------------------------------------------------------
#define Bn   16
#define Ln   96
#define Nn   512
#define Dn   512
#define Hn   8
#define En   64          // Dn / Hn
#define DFFn 2048
#define PREDn 96
#define Tn   516         // Nn + NMARK tokens
#define Mrows (Bn * Tn)  // 8256, multiple of 64

typedef __attribute__((ext_vector_type(16))) __bf16 bf16x16;
typedef __attribute__((ext_vector_type(8)))  float  f32x8;
typedef __attribute__((ext_vector_type(4)))  unsigned int u32x4;
typedef __attribute__((ext_vector_type(8)))  int i32x8;
typedef __attribute__((ext_vector_type(4)))  int i32x4;

// float -> bf16 (round-to-nearest-even)
__device__ __forceinline__ __bf16 f2bf(float f) {
  union { float f; unsigned u; } in; in.f = f;
  unsigned u = in.u;
  unsigned r = u + 0x7FFFu + ((u >> 16) & 1u);
  union { unsigned short s; __bf16 b; } out;
  out.s = (unsigned short)(r >> 16);
  return out.b;
}

// K-index inside a 16-bit 16x32 A/B WMMA fragment (ISA 7.12.2):
//   VGPR v = e/2 ; K = (v%4)*2 + (e&1) + 8*half + 16*(v/4)
__device__ __forceinline__ int frag_k(int e, int half) {
  int v = e >> 1;
  return ((v & 3) << 1) + (e & 1) + (half << 3) + ((v >> 2) << 4);
}

// ---------------------------------------------------------------------------
// TDM: 2-D bf16 tile load global -> LDS (ISA ch.8 D# layout).
//   d0/d1   : tensor extents (elements) measured from global_addr (OOB -> 0)
//   t0/t1   : tile dims (elements); stride0: row stride (elements)
// Issued by one wave; complete with s_wait_tensorcnt 0 + barrier.
// ---------------------------------------------------------------------------
__device__ __forceinline__ void tdm_load_2d(unsigned lds_off, const void* gptr,
                                            unsigned d0, unsigned d1,
                                            unsigned t0, unsigned t1,
                                            unsigned stride0) {
  unsigned long long ga = (unsigned long long)(uintptr_t)gptr;
  u32x4 g0;
  g0[0] = 1u;                                               // count=1, user mode
  g0[1] = lds_off;                                          // lds_addr (bytes)
  g0[2] = (unsigned)ga;                                     // global_addr[31:0]
  g0[3] = (unsigned)((ga >> 32) & 0x01FFFFFFu) | 0x80000000u; // addr[56:32] | type=2
  i32x8 g1;
  g1[0] = (int)(1u << 16);                                  // data_size = 2 bytes
  g1[1] = (int)(d0 << 16);                                  // tensor_dim0[15:0]
  g1[2] = (int)((d0 >> 16) | (d1 << 16));                   // dim0 hi | dim1 lo
  g1[3] = (int)((d1 >> 16) | (t0 << 16));                   // dim1 hi | tile_dim0
  g1[4] = (int)(t1 & 0xFFFFu);                              // tile_dim1 (tile_dim2=0)
  g1[5] = (int)stride0;                                     // tensor_dim0_stride[31:0]
  g1[6] = 0;                                                // stride0 hi | dim1_stride lo
  g1[7] = 0;
  i32x4 z4 = {0, 0, 0, 0};
#if defined(__clang_major__) && (__clang_major__ >= 23)
  i32x8 z8 = {0, 0, 0, 0, 0, 0, 0, 0};
  __builtin_amdgcn_tensor_load_to_lds(g0, g1, z4, z4, z8, 0);
#else
  __builtin_amdgcn_tensor_load_to_lds(g0, g1, z4, z4, 0);
#endif
}

__device__ __forceinline__ unsigned lds_off_of(const void* p) {
  return (unsigned)(uintptr_t)p;   // generic LDS pointer: low 32 bits = LDS byte offset
}

// ---------------------------------------------------------------------------
// 0) f32 -> bf16 conversion pass (weights once per launch, deterministic)
// ---------------------------------------------------------------------------
__global__ void cvt_bf16_kernel(const float* __restrict__ src, __bf16* __restrict__ dst,
                                int n) {
  int id = blockIdx.x * blockDim.x + threadIdx.x;
  if (id < n) dst[id] = f2bf(src[id]);
}

// ---------------------------------------------------------------------------
// 1) RevIN
// ---------------------------------------------------------------------------
__global__ void revin_kernel(const float* __restrict__ xe,
                             const float* __restrict__ gamma,
                             const float* __restrict__ beta,
                             float* __restrict__ xn,
                             float* __restrict__ meanb,
                             float* __restrict__ stdb,
                             float* __restrict__ stats) {
  int id = blockIdx.x * blockDim.x + threadIdx.x;
  if (id >= Bn * Nn) return;
  int b = id >> 9, n = id & (Nn - 1);
  size_t base = (size_t)b * Ln * Nn + n;
  float s = 0.f, q = 0.f;
  for (int l = 0; l < Ln; ++l) { float v = xe[base + (size_t)l * Nn]; s += v; q += v * v; }
  float mean = s * (1.f / Ln);
  float var  = q * (1.f / Ln) - mean * mean;
  float std  = sqrtf(fmaxf(var, 0.f) + 1e-5f);
  meanb[id] = mean; stdb[id] = std;
  float ga = gamma[n], be = beta[n], inv = 1.f / std;
  float s2 = 0.f, q2 = 0.f;
  for (int l = 0; l < Ln; ++l) {
    float v = (xe[base + (size_t)l * Nn] - mean) * inv * ga + be;
    xn[base + (size_t)l * Nn] = v;
    s2 += v; q2 += v * v;
  }
  float mf = s2 * (1.f / Ln);
  float vf = q2 * (1.f / Ln) - mf * mf;
  stats[id * 2 + 0] = mf;
  stats[id * 2 + 1] = sqrtf(fmaxf(vf, 0.f) + 1e-6f);
}

// ---------------------------------------------------------------------------
// 2) Trend MLP
// ---------------------------------------------------------------------------
__global__ void trend_mlp_kernel(const float* __restrict__ stats,
                                 const float* __restrict__ w1, const float* __restrict__ b1,
                                 const float* __restrict__ w2, const float* __restrict__ b2,
                                 float* __restrict__ tw) {
  int id = blockIdx.x * blockDim.x + threadIdx.x;
  if (id >= Bn * Nn) return;
  float s0 = stats[id * 2], s1 = stats[id * 2 + 1];
  float h[16];
#pragma unroll
  for (int j = 0; j < 16; ++j) {
    float v = s0 * w1[j] + s1 * w1[16 + j] + b1[j];
    h[j] = fmaxf(v, 0.f);
  }
  float o[4]; float mx = -3.0e38f;
#pragma unroll
  for (int k = 0; k < 4; ++k) {
    float v = b2[k];
#pragma unroll
    for (int j = 0; j < 16; ++j) v += h[j] * w2[j * 4 + k];
    o[k] = v; mx = fmaxf(mx, v);
  }
  float z = 0.f;
#pragma unroll
  for (int k = 0; k < 4; ++k) { o[k] = __expf(o[k] - mx); z += o[k]; }
  float iz = 1.f / z;
#pragma unroll
  for (int k = 0; k < 4; ++k) tw[id * 4 + k] = o[k] * iz;
}

// ---------------------------------------------------------------------------
// 3) Weighted moving-average trend -> [B,N,L], L2-normalized rows
// ---------------------------------------------------------------------------
__global__ void trend_nt_kernel(const float* __restrict__ xn,
                                const float* __restrict__ tw,
                                float* __restrict__ nt) {
  int id = blockIdx.x * blockDim.x + threadIdx.x;
  if (id >= Bn * Nn) return;
  int b = id >> 9, n = id & (Nn - 1);
  size_t base = (size_t)b * Ln * Nn + n;
  const int   W[4]  = {4, 8, 12, 24};
  const float iW[4] = {1.f / 4.f, 1.f / 8.f, 1.f / 12.f, 1.f / 24.f};
  float x0 = xn[base];
  float S[4]; float tww[4];
#pragma unroll
  for (int w = 0; w < 4; ++w) { S[w] = (float)W[w] * x0; tww[w] = tw[id * 4 + w]; }
  size_t nb = (size_t)id * Ln;
  float nsq = 0.f;
  for (int l = 0; l < Ln; ++l) {
    float xl = xn[base + (size_t)l * Nn];
    float tr = 0.f;
#pragma unroll
    for (int w = 0; w < 4; ++w) {
      int j = l - W[w]; if (j < 0) j = 0;
      S[w] += xl - xn[base + (size_t)j * Nn];
      tr += tww[w] * S[w] * iW[w];
    }
    nt[nb + l] = tr; nsq += tr * tr;
  }
  float inv = 1.f / fmaxf(sqrtf(nsq), 1e-12f);
  for (int l = 0; l < Ln; ++l) nt[nb + l] *= inv;
}

// ---------------------------------------------------------------------------
// 4) Guidance gate = sigmoid(nt @ nt^T)
// ---------------------------------------------------------------------------
__global__ void guidance_kernel(const float* __restrict__ nt, float* __restrict__ gate) {
  __shared__ float Ai[16][Ln];
  __shared__ float Aj[16][Ln];
  int b = blockIdx.z;
  int i0 = blockIdx.y * 16, j0 = blockIdx.x * 16;
  int tid = threadIdx.x;
  for (int i = tid; i < 16 * Ln; i += 256) {
    int r = i / Ln, k = i % Ln;
    Ai[r][k] = nt[((size_t)b * Nn + i0 + r) * Ln + k];
    Aj[r][k] = nt[((size_t)b * Nn + j0 + r) * Ln + k];
  }
  __syncthreads();
  int ty = tid >> 4, tx = tid & 15;
  float d = 0.f;
  for (int k = 0; k < Ln; ++k) d += Ai[ty][k] * Aj[tx][k];
  gate[((size_t)b * Nn + i0 + ty) * Nn + j0 + tx] = 1.f / (1.f + __expf(-d));
}

// ---------------------------------------------------------------------------
// 5) Token build (bf16 out): tok[b,t,l] = t<512 ? x_norm[b,l,t] : x_mark[b,l,t-512]
// ---------------------------------------------------------------------------
__global__ void tok_kernel(const float* __restrict__ xn, const float* __restrict__ xm,
                           __bf16* __restrict__ tok) {
  int id = blockIdx.x * blockDim.x + threadIdx.x;
  if (id >= Bn * Tn * Ln) return;
  int l = id % Ln;
  int t = (id / Ln) % Tn;
  int b = id / (Ln * Tn);
  float v = (t < Nn) ? xn[((size_t)b * Ln + l) * Nn + t]
                     : xm[((size_t)b * Ln + l) * 4 + (t - Nn)];
  tok[id] = f2bf(v);
}

// ---------------------------------------------------------------------------
// 6) bf16-WMMA GEMM with TDM tile staging.
//    C = act(A[M,K]bf16 @ W[K,N]bf16 + bias) (+ resid f32); dual f32/bf16 out.
//    256 thr = 8 waves (4 M x 2 N); tile 64x128; K-step 32 (all K % 32 == 0).
// ---------------------------------------------------------------------------
template <int ACT>  // 0 = none, 1 = exact GELU
__global__ void gemm_kernel(const __bf16* __restrict__ A, const __bf16* __restrict__ W,
                            const float* __restrict__ bias, const float* __restrict__ resid,
                            float* __restrict__ Cf, __bf16* __restrict__ Cbf,
                            int M, int K, int N) {
  const int tid  = threadIdx.x;
  const int wv_  = tid >> 5, lane = tid & 31;
  const int ln16 = lane & 15, half = lane >> 4;
  const int wm = wv_ & 3, wn = wv_ >> 2;
  const int m0 = blockIdx.y * 64;
  const int n0 = blockIdx.x * 128;

  __shared__ __bf16 As[64][32];
  __shared__ __bf16 Ws[32][128];

  const unsigned asOff = lds_off_of(&As[0][0]);
  const unsigned wsOff = lds_off_of(&Ws[0][0]);

  f32x8 acc[4];
#pragma unroll
  for (int t = 0; t < 4; ++t)
#pragma unroll
    for (int i = 0; i < 8; ++i) acc[t][i] = 0.f;

  for (int kk = 0; kk < K; kk += 32) {
    if (wv_ == 0) {
      // A tile: 64 rows x 32 cols, row stride K (no edges in M/K)
      tdm_load_2d(asOff, A + (size_t)m0 * K + kk, 32u, 64u, 32u, 64u, (unsigned)K);
      // W tile: 32 rows x 128 cols, row stride N; N-edge zero-filled by tensor_dim0
      tdm_load_2d(wsOff, W + (size_t)kk * N + n0, (unsigned)(N - n0), 32u, 128u, 32u,
                  (unsigned)N);
      __builtin_amdgcn_s_wait_tensorcnt((short)0);
    }
    __syncthreads();
    bf16x16 af;
#pragma unroll
    for (int e = 0; e < 16; ++e) af[e] = As[wm * 16 + ln16][frag_k(e, half)];
#pragma unroll
    for (int t = 0; t < 4; ++t) {
      bf16x16 bfv;
#pragma unroll
      for (int e = 0; e < 16; ++e) bfv[e] = Ws[frag_k(e, half)][wn * 64 + t * 16 + ln16];
      acc[t] = __builtin_amdgcn_wmma_f32_16x16x32_bf16(false, af, false, bfv,
                                                       (short)0, acc[t], false, false);
    }
    __syncthreads();
  }

#pragma unroll
  for (int t = 0; t < 4; ++t) {
    int ng = n0 + wn * 64 + t * 16 + ln16;
    if (ng >= N) continue;
    float bv = bias[ng];
#pragma unroll
    for (int rr = 0; rr < 8; ++rr) {
      int mg = m0 + wm * 16 + rr + 8 * half;
      float v = acc[t][rr] + bv;
      if (ACT == 1) v = 0.5f * v * (1.f + erff(v * 0.70710678118f));
      if (resid) v += resid[(size_t)mg * N + ng];
      if (Cf)  Cf[(size_t)mg * N + ng] = v;
      if (Cbf) Cbf[(size_t)mg * N + ng] = f2bf(v);
    }
  }
}

// ---------------------------------------------------------------------------
// 7) Fused attention per (b, h, 16-row tile): TDM-staged bf16 Q/K/V,
//    WMMA scores -> masked softmax -> guidance gate -> renorm -> WMMA ctx.
//    128 threads = 4 waves; TDM OOB rows (t >= 516) arrive as zeros.
// ---------------------------------------------------------------------------
__global__ void attn_kernel(const __bf16* __restrict__ qb, const __bf16* __restrict__ kb,
                            const __bf16* __restrict__ vb, const float* __restrict__ gate,
                            __bf16* __restrict__ ctx) {
  const int l0  = blockIdx.x * 16;
  const int h   = blockIdx.y;
  const int b   = blockIdx.z;
  const int tid = threadIdx.x;
  const int wv_ = tid >> 5, lane = tid & 31;
  const int ln16 = lane & 15, half = lane >> 4;

  __shared__ __bf16 Qs[16][64];      // 2 KB
  __shared__ __bf16 KVs[64][64];     // 8 KB  (K rows, later V rows)
  __shared__ float  Sf[16][528];     // 33 KB
  __shared__ __bf16 Ab[16][544];     // 17 KB
  __shared__ float  red[16][8];

  const size_t base = (size_t)b * Tn * Dn + (size_t)h * En;
  const unsigned qsOff = lds_off_of(&Qs[0][0]);
  const unsigned kvOff = lds_off_of(&KVs[0][0]);

  if (wv_ == 0) {
    // Q tile 16x64 (rows beyond T-l0 zero-filled)
    tdm_load_2d(qsOff, qb + base + (size_t)l0 * Dn, 64u, (unsigned)(Tn - l0),
                64u, 16u, (unsigned)Dn);
  }

  // ---- scores: S = (Q @ K^T) / 8 ; 4 col-tiles per staged K group ----
  for (int g = 0; g < 9; ++g) {
    if (wv_ == 0) {
      tdm_load_2d(kvOff, kb + base + (size_t)(g * 64) * Dn, 64u,
                  (unsigned)(Tn - g * 64), 64u, 64u, (unsigned)Dn);
      __builtin_amdgcn_s_wait_tensorcnt((short)0);
    }
    __syncthreads();
    int tile = g * 4 + wv_;
    if (tile < 33) {
      f32x8 acc;
#pragma unroll
      for (int i = 0; i < 8; ++i) acc[i] = 0.f;
#pragma unroll
      for (int kk = 0; kk < 64; kk += 32) {
        bf16x16 af, bfv;
#pragma unroll
        for (int e = 0; e < 16; ++e) {
          int kl = kk + frag_k(e, half);
          af[e]  = Qs[ln16][kl];
          bfv[e] = KVs[wv_ * 16 + ln16][kl];   // B[k][s] = K[s][k]
        }
        acc = __builtin_amdgcn_wmma_f32_16x16x32_bf16(false, af, false, bfv,
                                                      (short)0, acc, false, false);
      }
      int sG = tile * 16 + ln16;
#pragma unroll
      for (int r = 0; r < 8; ++r) Sf[r + 8 * half][sG] = acc[r] * 0.125f;
    }
    __syncthreads();
  }

  // mask padded cols [516,528)
  for (int i = tid; i < 16 * 12; i += 128) Sf[i / 12][Tn + i % 12] = -3.0e38f;
  __syncthreads();

  // ---- softmax + gate + renorm (8 threads per row) ----
  {
    int r = tid >> 3, sub = tid & 7;
    float mx = -3.0e38f;
    for (int s = sub; s < 528; s += 8) mx = fmaxf(mx, Sf[r][s]);
    red[r][sub] = mx; __syncthreads();
    mx = red[r][0];
#pragma unroll
    for (int i = 1; i < 8; ++i) mx = fmaxf(mx, red[r][i]);
    __syncthreads();

    float sm = 0.f;
    for (int s = sub; s < 528; s += 8) { float e = __expf(Sf[r][s] - mx); Sf[r][s] = e; sm += e; }
    red[r][sub] = sm; __syncthreads();
    sm = 0.f;
#pragma unroll
    for (int i = 0; i < 8; ++i) sm += red[r][i];
    __syncthreads();
    float invZ = 1.f / sm;

    int lg = l0 + r;
    float gs = 0.f;
    for (int s = sub; s < 528; s += 8) {
      float gv = (lg < Nn && s < Nn) ? gate[((size_t)b * Nn + lg) * Nn + s] : 0.5f;
      float v = Sf[r][s] * invZ * gv;
      Sf[r][s] = v; gs += v;
    }
    red[r][sub] = gs; __syncthreads();
    gs = 0.f;
#pragma unroll
    for (int i = 0; i < 8; ++i) gs += red[r][i];
    __syncthreads();
    float inv2 = 1.f / (gs + 1e-6f);
    for (int s = sub; s < 528; s += 8) Ab[r][s] = f2bf(Sf[r][s] * inv2);
    for (int s = 528 + sub; s < 544; s += 8) Ab[r][s] = f2bf(0.f);
  }
  __syncthreads();

  // ---- ctx = attn @ V ; each wave owns 16 of the 64 head-dim cols ----
  f32x8 acc;
#pragma unroll
  for (int i = 0; i < 8; ++i) acc[i] = 0.f;
  for (int kk = 0; kk < 544; kk += 32) {
    if (wv_ == 0) {
      unsigned rows_left = (kk < Tn) ? (unsigned)(Tn - kk) : 1u;  // OOB rows -> zeros
      tdm_load_2d(kvOff, vb + base + (size_t)kk * Dn, 64u, rows_left,
                  64u, 32u, (unsigned)Dn);
      __builtin_amdgcn_s_wait_tensorcnt((short)0);
    }
    __syncthreads();
    bf16x16 af, bfv;
#pragma unroll
    for (int e = 0; e < 16; ++e) {
      int kl = frag_k(e, half);
      af[e]  = Ab[ln16][kk + kl];
      bfv[e] = KVs[kl][wv_ * 16 + ln16];   // B[s][d] = V[s][d]
    }
    acc = __builtin_amdgcn_wmma_f32_16x16x32_bf16(false, af, false, bfv,
                                                  (short)0, acc, false, false);
    __syncthreads();
  }
#pragma unroll
  for (int rr = 0; rr < 8; ++rr) {
    int t = l0 + rr + 8 * half;
    if (t < Tn) ctx[base + (size_t)t * Dn + wv_ * 16 + ln16] = f2bf(acc[rr]);
  }
}

// ---------------------------------------------------------------------------
// 8) LayerNorm over D=512, in-place f32 + optional bf16 copy
// ---------------------------------------------------------------------------
__global__ void ln_kernel(float* __restrict__ x, __bf16* __restrict__ xbf,
                          const float* __restrict__ g, const float* __restrict__ b) {
  const int row = blockIdx.x, tid = threadIdx.x;
  __shared__ float rs[256], rq[256];
  size_t off = (size_t)row * Dn;
  float v0 = x[off + tid], v1 = x[off + 256 + tid];
  rs[tid] = v0 + v1; rq[tid] = v0 * v0 + v1 * v1;
  __syncthreads();
  for (int st = 128; st > 0; st >>= 1) {
    if (tid < st) { rs[tid] += rs[tid + st]; rq[tid] += rq[tid + st]; }
    __syncthreads();
  }
  float mean = rs[0] * (1.f / Dn);
  float var  = rq[0] * (1.f / Dn) - mean * mean;
  float inv  = rsqrtf(var + 1e-5f);
  float o0 = (v0 - mean) * inv * g[tid]       + b[tid];
  float o1 = (v1 - mean) * inv * g[tid + 256] + b[tid + 256];
  x[off + tid]       = o0;
  x[off + 256 + tid] = o1;
  if (xbf) {
    xbf[off + tid]       = f2bf(o0);
    xbf[off + 256 + tid] = f2bf(o1);
  }
}

// ---------------------------------------------------------------------------
// 9) Inverse RevIN + transpose
// ---------------------------------------------------------------------------
__global__ void out_kernel(const float* __restrict__ pout, const float* __restrict__ gamma,
                           const float* __restrict__ beta, const float* __restrict__ meanb,
                           const float* __restrict__ stdb, float* __restrict__ out) {
  int id = blockIdx.x * blockDim.x + threadIdx.x;
  if (id >= Bn * PREDn * Nn) return;
  int n = id & (Nn - 1);
  int p = (id >> 9) % PREDn;
  int b = id / (PREDn * Nn);
  float v = pout[((size_t)b * Tn + n) * PREDn + p];
  v = (v - beta[n]) / (gamma[n] + 1e-5f) * stdb[b * Nn + n] + meanb[b * Nn + n];
  out[id] = v;
}

// ---------------------------------------------------------------------------
// Launcher
// ---------------------------------------------------------------------------
static void launch_gemm(const __bf16* A, const __bf16* W, const float* bias,
                        const float* resid, float* Cf, __bf16* Cbf,
                        int M, int K, int N, int act, hipStream_t s) {
  dim3 grid((N + 127) / 128, M / 64), blk(256);
  if (act) gemm_kernel<1><<<grid, blk, 0, s>>>(A, W, bias, resid, Cf, Cbf, M, K, N);
  else     gemm_kernel<0><<<grid, blk, 0, s>>>(A, W, bias, resid, Cf, Cbf, M, K, N);
}

static void launch_cvt(const float* src, __bf16* dst, int n, hipStream_t s) {
  cvt_bf16_kernel<<<(n + 255) / 256, 256, 0, s>>>(src, dst, n);
}

extern "C" void kernel_launch(void* const* d_in, const int* in_sizes, int n_in,
                              void* d_out, int out_size, void* d_ws, size_t ws_size,
                              hipStream_t stream) {
  (void)in_sizes; (void)n_in; (void)out_size; (void)ws_size;
  const float* x_enc  = (const float*)d_in[0];
  const float* x_mark = (const float*)d_in[1];
  const float* gamma  = (const float*)d_in[4];
  const float* beta   = (const float*)d_in[5];
  const float* tp_w1  = (const float*)d_in[6];
  const float* tp_b1  = (const float*)d_in[7];
  const float* tp_w2  = (const float*)d_in[8];
  const float* tp_b2  = (const float*)d_in[9];
  const float* emb_w  = (const float*)d_in[10];
  const float* emb_b  = (const float*)d_in[11];
  const float* wq = (const float*)d_in[12]; const float* bq = (const float*)d_in[13];
  const float* wk = (const float*)d_in[14]; const float* bk = (const float*)d_in[15];
  const float* wv = (const float*)d_in[16]; const float* bv = (const float*)d_in[17];
  const float* wo = (const float*)d_in[18]; const float* bo = (const float*)d_in[19];
  const float* w1 = (const float*)d_in[20]; const float* b1 = (const float*)d_in[21];
  const float* w2 = (const float*)d_in[22]; const float* b2 = (const float*)d_in[23];
  const float* ln1g = (const float*)d_in[24]; const float* ln1b = (const float*)d_in[25];
  const float* ln2g = (const float*)d_in[26]; const float* ln2b = (const float*)d_in[27];
  const float* normg = (const float*)d_in[28]; const float* normb = (const float*)d_in[29];
  const float* proj_w = (const float*)d_in[30]; const float* proj_b = (const float*)d_in[31];
  float* out = (float*)d_out;

  // ---- workspace carve-up: f32 region then bf16 region (~159 MB total) ----
  float* fw = (float*)d_ws;
  float* xn    = fw; fw += (size_t)Bn * Ln * Nn;
  float* meanb = fw; fw += (size_t)Bn * Nn;
  float* stdb  = fw; fw += (size_t)Bn * Nn;
  float* stats = fw; fw += (size_t)Bn * Nn * 2;
  float* tw    = fw; fw += (size_t)Bn * Nn * 4;
  float* ntb   = fw; fw += (size_t)Bn * Nn * Ln;
  float* gateb = fw; fw += (size_t)Bn * Nn * Nn;
  float* xa    = fw; fw += (size_t)Mrows * Dn;
  float* xb    = fw; fw += (size_t)Mrows * Dn;
  float* pout  = fw; fw += (size_t)Mrows * PREDn;

  __bf16* bw = (__bf16*)fw;
  __bf16* tokbf = bw; bw += (size_t)Mrows * Ln;
  __bf16* xabf  = bw; bw += (size_t)Mrows * Dn;
  __bf16* xbbf  = bw; bw += (size_t)Mrows * Dn;
  __bf16* ctxbf = bw; bw += (size_t)Mrows * Dn;
  __bf16* qbf   = bw; bw += (size_t)Mrows * Dn;
  __bf16* kbf   = bw; bw += (size_t)Mrows * Dn;
  __bf16* vbf   = bw; bw += (size_t)Mrows * Dn;
  __bf16* hbbf  = bw; bw += (size_t)Mrows * DFFn;
  __bf16* embbf = bw; bw += (size_t)Ln * Dn;
  __bf16* wqbf  = bw; bw += (size_t)2 * Dn * Dn;
  __bf16* wkbf  = bw; bw += (size_t)2 * Dn * Dn;
  __bf16* wvbf  = bw; bw += (size_t)2 * Dn * Dn;
  __bf16* wobf  = bw; bw += (size_t)2 * Dn * Dn;
  __bf16* w1bf  = bw; bw += (size_t)2 * Dn * DFFn;
  __bf16* w2bf  = bw; bw += (size_t)2 * DFFn * Dn;
  __bf16* prjbf = bw; bw += (size_t)Dn * PREDn;

  // ---- weight conversion (once per launch; deterministic) ----
  launch_cvt(emb_w,  embbf, Ln * Dn, stream);
  launch_cvt(wq,     wqbf,  2 * Dn * Dn, stream);
  launch_cvt(wk,     wkbf,  2 * Dn * Dn, stream);
  launch_cvt(wv,     wvbf,  2 * Dn * Dn, stream);
  launch_cvt(wo,     wobf,  2 * Dn * Dn, stream);
  launch_cvt(w1,     w1bf,  2 * Dn * DFFn, stream);
  launch_cvt(w2,     w2bf,  2 * DFFn * Dn, stream);
  launch_cvt(proj_w, prjbf, Dn * PREDn, stream);

  // ---- preamble ----
  revin_kernel    <<<32, 256, 0, stream>>>(x_enc, gamma, beta, xn, meanb, stdb, stats);
  trend_mlp_kernel<<<32, 256, 0, stream>>>(stats, tp_w1, tp_b1, tp_w2, tp_b2, tw);
  trend_nt_kernel <<<32, 256, 0, stream>>>(xn, tw, ntb);
  guidance_kernel <<<dim3(32, 32, Bn), 256, 0, stream>>>(ntb, gateb);
  tok_kernel      <<<(Bn * Tn * Ln + 255) / 256, 256, 0, stream>>>(xn, x_mark, tokbf);

  // ---- inverted embedding: x = tok @ emb_w + emb_b (f32 for residual, bf16 for GEMMs)
  launch_gemm(tokbf, embbf, emb_b, nullptr, xa, xabf, Mrows, Ln, Dn, 0, stream);

  // ---- encoder layers ----
  for (int i = 0; i < 2; ++i) {
    const __bf16* Wq = wqbf + (size_t)i * Dn * Dn;   const float* Bq = bq + (size_t)i * Dn;
    const __bf16* Wk = wkbf + (size_t)i * Dn * Dn;   const float* Bk = bk + (size_t)i * Dn;
    const __bf16* Wv = wvbf + (size_t)i * Dn * Dn;   const float* Bv = bv + (size_t)i * Dn;
    const __bf16* Wo = wobf + (size_t)i * Dn * Dn;   const float* Bo = bo + (size_t)i * Dn;
    const __bf16* W1 = w1bf + (size_t)i * Dn * DFFn; const float* B1 = b1 + (size_t)i * DFFn;
    const __bf16* W2 = w2bf + (size_t)i * DFFn * Dn; const float* B2 = b2 + (size_t)i * Dn;

    launch_gemm(xabf, Wq, Bq, nullptr, nullptr, qbf, Mrows, Dn, Dn, 0, stream);
    launch_gemm(xabf, Wk, Bk, nullptr, nullptr, kbf, Mrows, Dn, Dn, 0, stream);
    launch_gemm(xabf, Wv, Bv, nullptr, nullptr, vbf, Mrows, Dn, Dn, 0, stream);

    attn_kernel<<<dim3(33, Hn, Bn), 128, 0, stream>>>(qbf, kbf, vbf, gateb, ctxbf);

    launch_gemm(ctxbf, Wo, Bo, xa, xb, nullptr, Mrows, Dn, Dn, 0, stream);    // +resid
    ln_kernel<<<Mrows, 256, 0, stream>>>(xb, xbbf, ln1g + (size_t)i * Dn,
                                         ln1b + (size_t)i * Dn);

    launch_gemm(xbbf, W1, B1, nullptr, nullptr, hbbf, Mrows, Dn, DFFn, 1, stream); // GELU
    launch_gemm(hbbf, W2, B2, xb, xa, nullptr, Mrows, DFFn, Dn, 0, stream);        // +resid
    ln_kernel<<<Mrows, 256, 0, stream>>>(xa, xabf, ln2g + (size_t)i * Dn,
                                         ln2b + (size_t)i * Dn);
  }

  // ---- head ----
  ln_kernel<<<Mrows, 256, 0, stream>>>(xa, xabf, normg, normb);
  launch_gemm(xabf, prjbf, proj_b, nullptr, pout, nullptr, Mrows, Dn, PREDn, 0, stream);
  out_kernel<<<(Bn * PREDn * Nn + 255) / 256, 256, 0, stream>>>(pout, gamma, beta,
                                                                meanb, stdb, out);
}
